// GATPSPGNNNL_635655160275
// MI455X (gfx1250) — compile-verified, matrix operand order checked by reading, not run on hardware
//
#include <hip/hip_runtime.h>
#include <hip/hip_bf16.h>
#include <stdint.h>

// ---------------------------------------------------------------------------
// GAT (3-layer) for MI455X / gfx1250.
//  - GEMMs: bf16 WMMA (v_wmma_f32_16x16x32_bf16), f32 accumulate.
//  - Edge softmax/aggregate: native f32 global atomics (add / max-key).
// ---------------------------------------------------------------------------

typedef __bf16 bf16_t;
typedef __attribute__((ext_vector_type(8)))  bf16_t v8bf;
typedef __attribute__((ext_vector_type(16))) bf16_t v16bf;
typedef __attribute__((ext_vector_type(8)))  float  v8f;

#define NEG_SLOPE 0.2f
#define COLS 256      // H*D == H*OUT == 256 for every layer
#define HEADS 4
#define DHEAD 64

// Build the 16-element bf16 WMMA operand from two contiguous 16-byte chunks
// (matches CDNA5 16-bit A/B operand layout: per lane, K {base..base+7} then
// {base+16..base+23}, with base = ksel*8).
static __device__ __forceinline__ v16bf ld_pair(const bf16_t* p0, const bf16_t* p1) {
  v8bf lo = *reinterpret_cast<const v8bf*>(p0);
  v8bf hi = *reinterpret_cast<const v8bf*>(p1);
  v16bf r;
#pragma unroll
  for (int i = 0; i < 8; ++i) { r[i] = lo[i]; r[i + 8] = hi[i]; }
  return r;
}

// C[M,256] = A[M,K] (bf16, row-major) * W[K,256]; Wt is W transposed: [256,K].
// One wave computes a 16(M) x 64(N) tile: 4 accumulators reuse one A operand.
__global__ void gemm_bf16_wmma(const bf16_t* __restrict__ A,
                               const bf16_t* __restrict__ Wt,
                               float* __restrict__ C,
                               int M, int K) {
  const int wave = (int)((blockIdx.x * blockDim.x + threadIdx.x) >> 5);
  const int lane = (int)(threadIdx.x & 31);
  const int mtiles = M >> 4;
  const int tm = wave >> 2;   // 16-row tile index
  const int tn = wave & 3;    // 64-col group index
  if (tm >= mtiles) return;

  const int lrow = lane & 15;
  const int ksel = lane >> 4;

  const bf16_t* arow = A + (size_t)((tm << 4) + lrow) * K;
  const int col0 = tn * 64 + lrow;
  const bf16_t* b0 = Wt + (size_t)(col0 +  0) * K;
  const bf16_t* b1 = Wt + (size_t)(col0 + 16) * K;
  const bf16_t* b2 = Wt + (size_t)(col0 + 32) * K;
  const bf16_t* b3 = Wt + (size_t)(col0 + 48) * K;

  v8f acc0 = {}, acc1 = {}, acc2 = {}, acc3 = {};
  for (int kb = 0; kb < K; kb += 32) {
    const int o0 = kb + ksel * 8;
    const int o1 = o0 + 16;
    v16bf a   = ld_pair(arow + o0, arow + o1);
    v16bf vb0 = ld_pair(b0 + o0, b0 + o1);
    v16bf vb1 = ld_pair(b1 + o0, b1 + o1);
    v16bf vb2 = ld_pair(b2 + o0, b2 + o1);
    v16bf vb3 = ld_pair(b3 + o0, b3 + o1);
    acc0 = __builtin_amdgcn_wmma_f32_16x16x32_bf16(false, a, false, vb0, (short)0, acc0, false, false);
    acc1 = __builtin_amdgcn_wmma_f32_16x16x32_bf16(false, a, false, vb1, (short)0, acc1, false, false);
    acc2 = __builtin_amdgcn_wmma_f32_16x16x32_bf16(false, a, false, vb2, (short)0, acc2, false, false);
    acc3 = __builtin_amdgcn_wmma_f32_16x16x32_bf16(false, a, false, vb3, (short)0, acc3, false, false);
  }

  // C/D layout: VGPR r, lanes 0-15 -> row r, col lane; lanes 16-31 -> row r+8.
  float* crow = C + (size_t)((tm << 4) + 8 * ksel) * COLS + tn * 64 + lrow;
#pragma unroll
  for (int r = 0; r < 8; ++r) {
    crow[(size_t)r * COLS +  0] = acc0[r];
    crow[(size_t)r * COLS + 16] = acc1[r];
    crow[(size_t)r * COLS + 32] = acc2[r];
    crow[(size_t)r * COLS + 48] = acc3[r];
  }
}

// W [K,256] f32 -> Wt [256,K] bf16
__global__ void k_transpose_w(const float* __restrict__ W, bf16_t* __restrict__ Wt, int K) {
  int idx = blockIdx.x * blockDim.x + threadIdx.x;
  if (idx >= K * COLS) return;
  int n = idx / K, k = idx - n * K;
  Wt[idx] = (bf16_t)W[(size_t)k * COLS + n];
}

// Layer-0 input: [fvs | pos_enc] -> bf16 [N,160]
__global__ void k_pack0(const float* __restrict__ fvs, const float* __restrict__ pos,
                        bf16_t* __restrict__ A, int N) {
  int idx = blockIdx.x * blockDim.x + threadIdx.x;
  if (idx >= N * 160) return;
  int n = idx / 160, c = idx - n * 160;
  float v = (c < 128) ? fvs[(size_t)n * 128 + c] : pos[(size_t)n * 32 + (c - 128)];
  A[idx] = (bf16_t)v;
}

// Mid-layer input: [elu(rst) | pos_enc] -> bf16 [N,288]
__global__ void k_elu_pack(const float* __restrict__ rst, const float* __restrict__ pos,
                           bf16_t* __restrict__ A, int N) {
  int idx = blockIdx.x * blockDim.x + threadIdx.x;
  if (idx >= N * 288) return;
  int n = idx / 288, c = idx - n * 288;
  float v;
  if (c < 256) { v = rst[(size_t)n * 256 + c]; v = (v > 0.f) ? v : (__expf(v) - 1.f); }
  else          v = pos[(size_t)n * 32 + (c - 256)];
  A[idx] = (bf16_t)v;
}

// el[n,h] = <feat[n,h,:], al[h,:]>; er likewise.
__global__ void k_logits(const float* __restrict__ feat, const float* __restrict__ al,
                         const float* __restrict__ ar, float* __restrict__ el,
                         float* __restrict__ er, int N) {
  int idx = blockIdx.x * blockDim.x + threadIdx.x;
  if (idx >= N * HEADS) return;
  int n = idx >> 2, hh = idx & 3;
  const float* f = feat + (size_t)n * COLS + hh * DHEAD;
  const float* a = al + hh * DHEAD;
  const float* b = ar + hh * DHEAD;
  float s0 = 0.f, s1 = 0.f;
#pragma unroll 8
  for (int d = 0; d < DHEAD; ++d) { float v = f[d]; s0 += v * a[d]; s1 += v * b[d]; }
  el[idx] = s0; er[idx] = s1;
}

__global__ void k_zero_u32(unsigned* __restrict__ p, int n) {
  int idx = blockIdx.x * blockDim.x + threadIdx.x;
  if (idx < n) p[idx] = 0u;
}

// monotone int key for f32 atomic max (0u == smallest possible key)
static __device__ __forceinline__ unsigned fkey(float f) {
  unsigned u = __float_as_uint(f);
  return (u & 0x80000000u) ? ~u : (u | 0x80000000u);
}

__global__ void k_edge_max(const int* __restrict__ src, const int* __restrict__ dst,
                           const float* __restrict__ el, const float* __restrict__ er,
                           unsigned* __restrict__ emax, int E) {
  int idx = blockIdx.x * blockDim.x + threadIdx.x;
  if (idx >= E * HEADS) return;
  int e = idx >> 2, hh = idx & 3;
  int s = src[e], d = dst[e];
  float x = el[s * HEADS + hh] + er[d * HEADS + hh];
  x = (x >= 0.f) ? x : NEG_SLOPE * x;
  atomicMax(emax + (size_t)d * HEADS + hh, fkey(x));
}

// keys -> f32 max in place; also zero denom (same [N*H] extent)
__global__ void k_decode_max(unsigned* __restrict__ emax, float* __restrict__ denom, int n4) {
  int idx = blockIdx.x * blockDim.x + threadIdx.x;
  if (idx >= n4) return;
  unsigned k = emax[idx];
  float f = (k & 0x80000000u) ? __uint_as_float(k & 0x7FFFFFFFu) : __uint_as_float(~k);
  reinterpret_cast<float*>(emax)[idx] = f;
  denom[idx] = 0.f;
}

__global__ void k_edge_exp(const int* __restrict__ src, const int* __restrict__ dst,
                           const float* __restrict__ el, const float* __restrict__ er,
                           const float* __restrict__ emaxf, float* __restrict__ denom,
                           float* __restrict__ alpha, int E) {
  int idx = blockIdx.x * blockDim.x + threadIdx.x;
  if (idx >= E * HEADS) return;
  int e = idx >> 2, hh = idx & 3;
  int s = src[e], d = dst[e];
  float x = el[s * HEADS + hh] + er[d * HEADS + hh];
  x = (x >= 0.f) ? x : NEG_SLOPE * x;
  float ee = __expf(x - emaxf[(size_t)d * HEADS + hh]);
  __hip_atomic_fetch_add(denom + (size_t)d * HEADS + hh, ee,
                         __ATOMIC_RELAXED, __HIP_MEMORY_SCOPE_AGENT);
  alpha[idx] = ee;
}

__global__ void k_alpha_norm(const int* __restrict__ dst, const float* __restrict__ denom,
                             float* __restrict__ alpha, int E) {
  int idx = blockIdx.x * blockDim.x + threadIdx.x;
  if (idx >= E * HEADS) return;
  int e = idx >> 2, hh = idx & 3;
  alpha[idx] = alpha[idx] / denom[(size_t)dst[e] * HEADS + hh];
}

// One block per edge; 256 contiguous channels; native f32 scatter-add.
__global__ void k_edge_aggregate(const int* __restrict__ src, const int* __restrict__ dst,
                                 const float* __restrict__ feat, const float* __restrict__ alpha,
                                 float* __restrict__ rst, int E) {
  int e = blockIdx.x;
  int c = threadIdx.x;
  int s = src[e], d = dst[e];
  float a = alpha[(size_t)e * HEADS + (c >> 6)];
  float m = feat[(size_t)s * COLS + c] * a;
  __hip_atomic_fetch_add(rst + (size_t)d * COLS + c, m,
                         __ATOMIC_RELAXED, __HIP_MEMORY_SCOPE_AGENT);
}

// Final: elu then mean over the 4 heads -> [N,64]
__global__ void k_final_mean(const float* __restrict__ rst, float* __restrict__ out, int N) {
  int idx = blockIdx.x * blockDim.x + threadIdx.x;
  if (idx >= N * 64) return;
  int n = idx >> 6, c = idx & 63;
  const float* r = rst + (size_t)n * COLS + c;
  float s = 0.f;
#pragma unroll
  for (int hh = 0; hh < HEADS; ++hh) {
    float v = r[hh * DHEAD];
    s += (v > 0.f) ? v : (__expf(v) - 1.f);
  }
  out[idx] = s * 0.25f;
}

__global__ void k_copy_f32(const float* __restrict__ s, float* __restrict__ d, int n) {
  int idx = blockIdx.x * blockDim.x + threadIdx.x;
  if (idx < n) d[idx] = s[idx];
}

static inline int cdiv_i(long long a, long long b) { return (int)((a + b - 1) / b); }

extern "C" void kernel_launch(void* const* d_in, const int* in_sizes, int n_in,
                              void* d_out, int out_size, void* d_ws, size_t ws_size,
                              hipStream_t stream) {
  const float* fvs = (const float*)d_in[0];
  const float* pos = (const float*)d_in[1];
  const int*   src = (const int*)d_in[2];
  const int*   dst = (const int*)d_in[3];
  const float* W[3]  = {(const float*)d_in[4],  (const float*)d_in[8],  (const float*)d_in[12]};
  const float* al[3] = {(const float*)d_in[5],  (const float*)d_in[9],  (const float*)d_in[13]};
  const float* ar[3] = {(const float*)d_in[6],  (const float*)d_in[10], (const float*)d_in[14]};
  const float* rW[3] = {(const float*)d_in[7],  (const float*)d_in[11], (const float*)d_in[15]};
  const int N = in_sizes[0] / 128;   // 50000 (multiple of 16)
  const int E = in_sizes[2];         // 800000
  const int Ks[3] = {160, 288, 288};

  // --- carve workspace (256B-aligned slices) ---
  char* p = (char*)d_ws;
  auto take = [&](size_t bytes) -> void* {
    char* r = p; p += (bytes + 255) & ~(size_t)255; return (void*)r;
  };
  bf16_t*   Abf   = (bf16_t*)take((size_t)N * 288 * sizeof(bf16_t));
  bf16_t*   WtA   = (bf16_t*)take((size_t)288 * COLS * sizeof(bf16_t));
  bf16_t*   WtR   = (bf16_t*)take((size_t)288 * COLS * sizeof(bf16_t));
  float*    feat  = (float*)take((size_t)N * COLS * sizeof(float));
  float*    rst   = (float*)take((size_t)N * COLS * sizeof(float));
  float*    el    = (float*)take((size_t)N * HEADS * sizeof(float));
  float*    er    = (float*)take((size_t)N * HEADS * sizeof(float));
  unsigned* emax  = (unsigned*)take((size_t)N * HEADS * sizeof(unsigned));
  float*    denom = (float*)take((size_t)N * HEADS * sizeof(float));
  float*    alpha = (float*)take((size_t)E * HEADS * sizeof(float));
  (void)ws_size; (void)n_in; (void)out_size;

  const int TB = 256;

  // layer-0 input: concat(fvs, pos) -> bf16
  k_pack0<<<cdiv_i((long long)N * 160, TB), TB, 0, stream>>>(fvs, pos, Abf, N);

  for (int L = 0; L < 3; ++L) {
    const int K = Ks[L];
    const int mwaves = (N / 16) * 4;           // one wave per 16x64 tile
    if (L > 0)
      k_elu_pack<<<cdiv_i((long long)N * 288, TB), TB, 0, stream>>>(rst, pos, Abf, N);

    k_transpose_w<<<cdiv_i((long long)K * COLS, TB), TB, 0, stream>>>(W[L],  WtA, K);
    k_transpose_w<<<cdiv_i((long long)K * COLS, TB), TB, 0, stream>>>(rW[L], WtR, K);

    gemm_bf16_wmma<<<cdiv_i(mwaves, 8), TB, 0, stream>>>(Abf, WtA, feat, N, K);
    // residual projection written straight into rst: edge atomics stack on top
    gemm_bf16_wmma<<<cdiv_i(mwaves, 8), TB, 0, stream>>>(Abf, WtR, rst, N, K);

    k_logits<<<cdiv_i((long long)N * HEADS, TB), TB, 0, stream>>>(feat, al[L], ar[L], el, er, N);
    k_zero_u32<<<cdiv_i((long long)N * HEADS, TB), TB, 0, stream>>>(emax, N * HEADS);
    k_edge_max<<<cdiv_i((long long)E * HEADS, TB), TB, 0, stream>>>(src, dst, el, er, emax, E);
    k_decode_max<<<cdiv_i((long long)N * HEADS, TB), TB, 0, stream>>>(emax, denom, N * HEADS);
    k_edge_exp<<<cdiv_i((long long)E * HEADS, TB), TB, 0, stream>>>(
        src, dst, el, er, (const float*)emax, denom, alpha, E);
    k_alpha_norm<<<cdiv_i((long long)E * HEADS, TB), TB, 0, stream>>>(dst, denom, alpha, E);
    k_edge_aggregate<<<E, COLS, 0, stream>>>(src, dst, feat, alpha, rst, E);
  }

  float* outp = (float*)d_out;
  k_final_mean<<<cdiv_i((long long)N * 64, TB), TB, 0, stream>>>(rst, outp, N);
  k_copy_f32<<<cdiv_i((long long)N * 32, TB), TB, 0, stream>>>(pos, outp + (size_t)N * 64, N * 32);
}
